// TextEmbedding_30958124269957
// MI455X (gfx1250) — matrix operand order repaired and checked by reading, not compiled, not run on hardware
//
#include <hip/hip_runtime.h>
#include <cstdint>

typedef __attribute__((ext_vector_type(16))) _Float16 v16h;
typedef __attribute__((ext_vector_type(8)))  _Float16 v8h;
typedef __attribute__((ext_vector_type(8)))  float    v8f;

#define B_   8
#define T_   256
#define C_   16
#define D_   512
#define TD_  1024
#define SEQ_ 4096
#define EPS_ 1e-6f

__device__ __forceinline__ float silu_f(float x) { return x / (1.0f + __expf(-x)); }

// ---------------------------------------------------------------------------
// Kernel 0: fill entire output with filler row (text_table row 0), float4-wide
// ---------------------------------------------------------------------------
__global__ void fill_filler(const float* __restrict__ text_table, float* __restrict__ out) {
    size_t idx = (size_t)blockIdx.x * blockDim.x + threadIdx.x;       // float4 index
    const float4* filler4 = (const float4*)text_table;                // row 0 = filler
    ((float4*)out)[idx] = filler4[idx & (D_ / 4 - 1)];                // 128 float4 per row
}

// ---------------------------------------------------------------------------
// Kernel P: per-batch exclusive prefix sum over char mask -> dest slot or -1
// ---------------------------------------------------------------------------
__global__ void scan_targets(const int* __restrict__ mask, int* __restrict__ tgt) {
    __shared__ int cnt[256];
    int b = blockIdx.x, tid = threadIdx.x;
    int base = b * (T_ * C_) + tid * 16;
    int c = 0;
    #pragma unroll
    for (int j = 0; j < 16; ++j) c += (mask[base + j] > 0) ? 1 : 0;
    cnt[tid] = c;
    __syncthreads();
    if (tid == 0) {
        int run = 0;
        for (int i = 0; i < 256; ++i) { int t = cnt[i]; cnt[i] = run; run += t; }
    }
    __syncthreads();
    int run = cnt[tid];
    for (int j = 0; j < 16; ++j) {
        int t = -1;
        if (mask[base + j] > 0) { if (run < SEQ_) t = run; run++; }
        tgt[base + j] = t;
    }
}

// ---------------------------------------------------------------------------
// Kernel W: convert f32 row-major [K][N] -> f16 column-major [N][K]
// (one-time prep so B-fragments become two contiguous 16-byte loads per lane)
// ---------------------------------------------------------------------------
__global__ void conv_transpose_f16(const float* __restrict__ in, _Float16* __restrict__ out,
                                   int K, int N) {
    int idx = blockIdx.x * blockDim.x + threadIdx.x;    // over K*N, coalesced read
    int k = idx / N, n = idx - k * N;
    out[(size_t)n * K + k] = (_Float16)in[idx];
}

// ---------------------------------------------------------------------------
// f16 WMMA fragment helpers (layouts per CDNA5 ISA 7.12.2, 16-bit matrices)
// A 16x32: lane<16 -> K {0..7, 16..23}; lane>=16 -> K {8..15, 24..31}
// B 32x16: lane<16 -> K {0..7, 16..23} of column N=lane; lane>=16 -> K {8..15, 24..31}
// ---------------------------------------------------------------------------
__device__ __forceinline__ v16h load_a_f16(const _Float16* __restrict__ row, int kbase, int koffA) {
    v8h lo = *(const v8h*)(row + kbase + koffA);
    v8h hi = *(const v8h*)(row + kbase + 16 + koffA);
    v16h a;
    #pragma unroll
    for (int j = 0; j < 8; ++j) { a[j] = lo[j]; a[8 + j] = hi[j]; }
    return a;
}
__device__ __forceinline__ v16h load_b_f16(const _Float16* __restrict__ col) {
    v8h lo = *(const v8h*)(col);
    v8h hi = *(const v8h*)(col + 16);
    v16h b;
    #pragma unroll
    for (int j = 0; j < 8; ++j) { b[j] = lo[j]; b[8 + j] = hi[j]; }
    return b;
}

// ---------------------------------------------------------------------------
// Kernel 1: token branch. 16 tokens/block, 128 threads (4 wave32).
// RMSNorm -> [16x1024]@[1024x512] via v_wmma_f32_16x16x32_f16 (f32 accum)
// -> RMSNorm -> SiLU -> W_proj -> (scale, shift, sigmoid(gate)) to workspace.
// ---------------------------------------------------------------------------
__global__ void __launch_bounds__(128)
token_branch(const int* __restrict__ token_ids,
             const float* __restrict__ token_table,
             const float* __restrict__ w_pre,
             const float* __restrict__ w_tok,
             const _Float16* __restrict__ Wdt,     // [N=512][K=1024] f16
             const float* __restrict__ W_proj,
             float* __restrict__ params) {
    __shared__ __align__(16) unsigned char smem[16 * TD_ * 2];   // 32 KB, reused
    _Float16 (*lds_tok)[TD_] = (_Float16 (*)[TD_])smem;          // f16 normalized tokens
    float    (*lds_f)[D_]    = (float (*)[D_])smem;              // f32 feats (after GEMM)
    __shared__ float red[16][8];
    __shared__ float rnorm[16];
    __shared__ float rn2[16];
    __shared__ float pp[16][8][3];
    __shared__ int   ids[16];

    int tid  = threadIdx.x;
    int lane = tid & 31;
    int wave = tid >> 5;
    int base = blockIdx.x * 16;

    if (tid < 16) ids[tid] = token_ids[base + tid];
    __syncthreads();

    // stage raw token rows as f16
    for (int i = tid; i < 16 * TD_; i += 128) {
        int m = i >> 10, k = i & (TD_ - 1);
        lds_tok[m][k] = (_Float16)token_table[(size_t)ids[m] * TD_ + k];
    }
    __syncthreads();
    {   // per-row sum of squares
        int m = tid >> 3, l8 = tid & 7;
        float s = 0.f;
        for (int k = l8; k < TD_; k += 8) { float x = (float)lds_tok[m][k]; s += x * x; }
        red[m][l8] = s;
    }
    __syncthreads();
    if (tid < 16) {
        float s = 0.f;
        #pragma unroll
        for (int j = 0; j < 8; ++j) s += red[tid][j];
        rnorm[tid] = rsqrtf(s * (1.0f / TD_) + EPS_);
    }
    __syncthreads();
    // normalize in place: f16( x * rnorm * w_pre )
    for (int i = tid; i < 16 * TD_; i += 128) {
        int m = i >> 10, k = i & (TD_ - 1);
        lds_tok[m][k] = (_Float16)((float)lds_tok[m][k] * rnorm[m] * w_pre[k]);
    }
    __syncthreads();

    v8f acc[8] = {};
    int mrow  = lane & 15;
    int koffA = (lane < 16) ? 0 : 8;
    int koffB = (lane < 16) ? 0 : 8;

    for (int kc = 0; kc < TD_ / 32; ++kc) {           // 32 K-chunks of 32
        int kbase = kc * 32;
        v16h a = load_a_f16(&lds_tok[mrow][0], kbase, koffA);
        #pragma unroll
        for (int i = 0; i < 8; ++i) {
            int n = (wave * 8 + i) * 16 + mrow;
            v16h b = load_b_f16(Wdt + (size_t)n * TD_ + kbase + koffB);
            acc[i] = __builtin_amdgcn_wmma_f32_16x16x32_f16(
                false, a, false, b, (short)0, acc[i], false, false);
        }
    }
    __syncthreads();                                   // GEMM done, reuse smem for feats

    {   // spill feats (C/D layout: vgpr r -> M=r lanes 0-15, M=8+r lanes 16-31)
        int mb = (lane < 16) ? 0 : 8;
        #pragma unroll
        for (int i = 0; i < 8; ++i) {
            int n = (wave * 8 + i) * 16 + mrow;
            #pragma unroll
            for (int r = 0; r < 8; ++r) lds_f[mb + r][n] = acc[i][r];
        }
    }
    __syncthreads();
    {   // RMSNorm of feats
        int m = tid >> 3, l8 = tid & 7;
        float s = 0.f;
        for (int n = l8; n < D_; n += 8) { float x = lds_f[m][n]; s += x * x; }
        red[m][l8] = s;
    }
    __syncthreads();
    if (tid < 16) {
        float s = 0.f;
        #pragma unroll
        for (int j = 0; j < 8; ++j) s += red[tid][j];
        rn2[tid] = rsqrtf(s * (1.0f / D_) + EPS_);
    }
    __syncthreads();
    {   // SiLU + W_proj (512x3) partial dots
        int m = tid >> 3, l8 = tid & 7;
        float r = rn2[m];
        float p0 = 0.f, p1 = 0.f, p2 = 0.f;
        for (int n = l8; n < D_; n += 8) {
            float f = lds_f[m][n] * r * w_tok[n];
            float s = silu_f(f);
            p0 += s * W_proj[n * 3 + 0];
            p1 += s * W_proj[n * 3 + 1];
            p2 += s * W_proj[n * 3 + 2];
        }
        pp[m][l8][0] = p0; pp[m][l8][1] = p1; pp[m][l8][2] = p2;
    }
    __syncthreads();
    if (tid < 16) {
        float p0 = 0.f, p1 = 0.f, p2 = 0.f;
        #pragma unroll
        for (int j = 0; j < 8; ++j) {
            p0 += pp[tid][j][0]; p1 += pp[tid][j][1]; p2 += pp[tid][j][2];
        }
        float gate = 1.0f / (1.0f + __expf(-p2));
        params[(base + tid) * 3 + 0] = p0;    // scale
        params[(base + tid) * 3 + 1] = p1;    // shift
        params[(base + tid) * 3 + 2] = gate;  // sigmoid already applied
    }
}

// ---------------------------------------------------------------------------
// Kernel 2: char branch. 4 (b,t) groups per block -> M=64 rows, 256 threads
// (8 wave32: mtile = wave&3, N-half = wave>>2, 16 N-tiles each, 128 acc VGPRs).
// RMSNorm (LDS f16) -> A = f16(silu(cn*(1+scale)+shift)) on the fly ->
// [64x512]@[512x512] via v_wmma_f32_16x16x32_f16 -> gated blend -> scatter.
// W_fusion re-read per block drops to 512 KB f16 (256 MB total from L2).
// ---------------------------------------------------------------------------
__global__ void __launch_bounds__(256)
char_branch(const int* __restrict__ char_ids,
            const float* __restrict__ text_table,
            const float* __restrict__ w_char,
            const _Float16* __restrict__ Wft,      // [N=512][K=512] f16
            const float* __restrict__ params,
            const int* __restrict__ tgt,
            float* __restrict__ out) {
    __shared__ __align__(16) _Float16 lds_cn[64][D_];   // 64 KB normalized chars
    __shared__ float red[64][4];
    __shared__ float rnorm[64];
    __shared__ int   cid[64];
    __shared__ int   ctgt[64];
    __shared__ float pgrp[4][3];

    int tid  = threadIdx.x;
    int lane = tid & 31;
    int wave = tid >> 5;
    int bt0  = blockIdx.x * 4;                     // 4 token groups per block

    if (tid < 64) {
        cid[tid]  = char_ids[bt0 * C_ + tid] + 1;  // ids shifted by +1
        ctgt[tid] = tgt[bt0 * C_ + tid];
    }
    if (tid < 12) pgrp[tid / 3][tid % 3] = params[(bt0 + tid / 3) * 3 + (tid % 3)];
    __syncthreads();

    // stage raw char embeddings as f16
    for (int i = tid; i < 64 * D_; i += 256) {
        int m = i >> 9, n = i & (D_ - 1);
        lds_cn[m][n] = (_Float16)text_table[(size_t)cid[m] * D_ + n];
    }
    __syncthreads();
    {   // per-row sum of squares (4 threads / row)
        int m = tid >> 2, l4 = tid & 3;
        float s = 0.f;
        for (int n = l4; n < D_; n += 4) { float x = (float)lds_cn[m][n]; s += x * x; }
        red[m][l4] = s;
    }
    __syncthreads();
    if (tid < 64) {
        float s = red[tid][0] + red[tid][1] + red[tid][2] + red[tid][3];
        rnorm[tid] = rsqrtf(s * (1.0f / D_) + EPS_);
    }
    __syncthreads();
    // normalize in place: cn_norm = f16( w_char * x * rnorm )
    for (int i = tid; i < 64 * D_; i += 256) {
        int m = i >> 9, n = i & (D_ - 1);
        lds_cn[m][n] = (_Float16)((float)lds_cn[m][n] * rnorm[m] * w_char[n]);
    }
    __syncthreads();

    int mtile = wave & 3;                          // which (b,t) group / M-tile
    int nhalf = wave >> 2;                         // N tiles [nhalf*16 .. +15]
    int mrow  = mtile * 16 + (lane & 15);
    int koffA = (lane < 16) ? 0 : 8;
    int koffB = (lane < 16) ? 0 : 8;
    float p_scale = pgrp[mtile][0];
    float p_shift = pgrp[mtile][1];
    float p_gate  = pgrp[mtile][2];
    float onescale = 1.0f + p_scale;

    v8f acc[16] = {};

    for (int kc = 0; kc < D_ / 32; ++kc) {         // 16 K-chunks of 32
        int kbase = kc * 32;
        const _Float16* crow = &lds_cn[mrow][0];
        v16h a;                                     // FiLM + SiLU fused A-fragment
        #pragma unroll
        for (int j = 0; j < 8; ++j) {
            float x0 = (float)crow[kbase + koffA + j];
            float x1 = (float)crow[kbase + 16 + koffA + j];
            a[j]     = (_Float16)silu_f(x0 * onescale + p_shift);
            a[8 + j] = (_Float16)silu_f(x1 * onescale + p_shift);
        }
        #pragma unroll
        for (int i = 0; i < 16; ++i) {
            int n = (nhalf * 16 + i) * 16 + (lane & 15);
            v16h b = load_b_f16(Wft + (size_t)n * D_ + kbase + koffB);
            acc[i] = __builtin_amdgcn_wmma_f32_16x16x32_f16(
                false, a, false, b, (short)0, acc[i], false, false);
        }
    }

    // gated blend + ragged scatter (skip invalid/overflow rows; filler pre-filled)
    int bt_g = bt0 + mtile;
    int b    = bt_g >> 8;                          // T_ == 256
    int mb   = (lane < 16) ? 0 : 8;
    float og = 1.0f - p_gate;
    #pragma unroll
    for (int i = 0; i < 16; ++i) {
        int n = (nhalf * 16 + i) * 16 + (lane & 15);
        #pragma unroll
        for (int r = 0; r < 8; ++r) {
            int m = mtile * 16 + mb + r;
            int t = ctgt[m];
            if (t >= 0) {
                float v = p_gate * acc[i][r] + og * (float)lds_cn[m][n];
                out[((size_t)b * SEQ_ + t) * D_ + n] = v;
            }
        }
    }
}

// ---------------------------------------------------------------------------
extern "C" void kernel_launch(void* const* d_in, const int* in_sizes, int n_in,
                              void* d_out, int out_size, void* d_ws, size_t ws_size,
                              hipStream_t stream) {
    const int*   token_ids   = (const int*)d_in[0];
    // d_in[1] token_ids_mask: unused by the reference
    const int*   char_ids    = (const int*)d_in[2];
    const int*   char_mask   = (const int*)d_in[3];
    // d_in[4] seq_len == 4096 (fixed by setup)
    const float* text_table  = (const float*)d_in[5];
    const float* token_table = (const float*)d_in[6];
    const float* w_pre       = (const float*)d_in[7];
    const float* w_tok       = (const float*)d_in[8];
    const float* w_char      = (const float*)d_in[9];
    const float* W_down      = (const float*)d_in[10];
    const float* W_proj      = (const float*)d_in[11];
    const float* W_fusion    = (const float*)d_in[12];
    float* out = (float*)d_out;

    // workspace layout (bytes):
    //   [      0,  24576) params  : B*T*3 f32
    //   [  32768, 163840) tgt     : B*T*C int
    //   [ 163840, 688128) Wft     : W_fusion^T f16 [512][512]
    //   [ 688128,1736704) Wdt     : W_down^T  f16 [512][1024]
    float*    params = (float*)d_ws;
    int*      tgtbuf = (int*)((char*)d_ws + 32768);
    _Float16* Wft    = (_Float16*)((char*)d_ws + 163840);
    _Float16* Wdt    = (_Float16*)((char*)d_ws + 688128);

    size_t total4 = (size_t)B_ * SEQ_ * D_ / 4;   // 4M float4 stores
    fill_filler <<<dim3((unsigned)(total4 / 256)), dim3(256), 0, stream>>>(text_table, out);
    scan_targets<<<dim3(B_),                      dim3(256), 0, stream>>>(char_mask, tgtbuf);
    conv_transpose_f16<<<dim3(TD_ * D_ / 256), dim3(256), 0, stream>>>(W_down,   Wdt, TD_, D_);
    conv_transpose_f16<<<dim3(D_  * D_ / 256), dim3(256), 0, stream>>>(W_fusion, Wft, D_,  D_);
    token_branch<<<dim3(B_ * T_ / 16), dim3(128), 0, stream>>>(
        token_ids, token_table, w_pre, w_tok, Wdt, W_proj, params);
    char_branch <<<dim3(B_ * T_ / 4),  dim3(256), 0, stream>>>(
        char_ids, text_table, w_char, Wft, params, tgtbuf, out);
}
